// GQAAttention_58566174048666
// MI455X (gfx1250) — compile-verified, matrix operand order checked by reading, not run on hardware
//
#include <hip/hip_runtime.h>

#define B_    2
#define T_    2048
#define C_    2048
#define NH_   16
#define NKV_  4
#define HD_   128
#define KVD_  (NKV_ * HD_)   // 512

typedef __attribute__((ext_vector_type(16))) __bf16 v16bf;
typedef __attribute__((ext_vector_type(8)))  float  v8f;
typedef unsigned short u16;
typedef unsigned int   u32;

__device__ __forceinline__ u16 f2bf(float f) {
  u32 x = __float_as_uint(f);
  u32 r = (x + 0x7FFFu + ((x >> 16) & 1u)) >> 16;   // round-to-nearest-even
  return (u16)r;
}
__device__ __forceinline__ float bf2f(u16 h) {
  return __uint_as_float(((u32)h) << 16);
}

union FragU { v16bf v; uint4 q[2]; };

// Load a 16x32 bf16 A/B-style WMMA fragment from a row-major LDS tile.
// ISA layout (16-bit A 16x32): lane L -> row = (L&15), k chunks at
// (L>>4)*8 .. +7  and  16+(L>>4)*8 .. +7  (two contiguous 16B chunks).
__device__ __forceinline__ v16bf frag_ld(const u16* tile, int pitch,
                                         int row0, int k0, int lane) {
  int r = row0 + (lane & 15);
  int g = (lane >> 4) << 3;
  const u16* p = tile + r * pitch + k0 + g;
  FragU f;
  f.q[0] = *(const uint4*)(p);
  f.q[1] = *(const uint4*)(p + 16);
  return f.v;
}

// Async 64-byte global -> LDS copy (per-lane). INST_OFFSET is added to both
// the LDS and global addresses, so consecutive offsets copy contiguously.
__device__ __forceinline__ void async_copy64(u32 lds_off, const void* gptr) {
  asm volatile(
      "global_load_async_to_lds_b128 %0, %1, off\n\t"
      "global_load_async_to_lds_b128 %0, %1, off offset:16\n\t"
      "global_load_async_to_lds_b128 %0, %1, off offset:32\n\t"
      "global_load_async_to_lds_b128 %0, %1, off offset:48"
      :: "v"(lds_off), "v"(gptr) : "memory");
}
__device__ __forceinline__ void async_copy32(u32 lds_off, const void* gptr) {
  asm volatile(
      "global_load_async_to_lds_b128 %0, %1, off\n\t"
      "global_load_async_to_lds_b128 %0, %1, off offset:16"
      :: "v"(lds_off), "v"(gptr) : "memory");
}
__device__ __forceinline__ void async_wait0() {
  asm volatile("s_wait_asynccnt 0" ::: "memory");
}
__device__ __forceinline__ void async_wait2() {   // allow 2 in flight (next tile)
  asm volatile("s_wait_asynccnt 2" ::: "memory");
}
__device__ __forceinline__ u32 lds_off32(const void* p) {
  return (u32)(uintptr_t)p;   // low 32 bits of generic addr == LDS offset
}

// ---------------------------------------------------------------- convert
__global__ void cvt_f32_bf16(const float4* __restrict__ src,
                             uint2* __restrict__ dst, int n4) {
  int i = blockIdx.x * blockDim.x + threadIdx.x;
  if (i < n4) {
    float4 f = src[i];
    uint2 o;
    o.x = (u32)f2bf(f.x) | ((u32)f2bf(f.y) << 16);
    o.y = (u32)f2bf(f.z) | ((u32)f2bf(f.w) << 16);
    dst[i] = o;
  }
}

// ---------------------------------------------------------------- RoPE
// In-place rotate-half RoPE on bf16 [B*T, NHEADS*HD] rows.
template <int NHEADS>
__global__ void rope_kernel(u16* __restrict__ q) {
  int idx = blockIdx.x * blockDim.x + threadIdx.x;
  if (idx >= B_ * T_ * NHEADS * (HD_ / 2)) return;
  int d = idx & 63;
  int h = (idx >> 6) & (NHEADS - 1);
  int r = idx / (64 * NHEADS);          // b*T + t  (power-of-two divide)
  int t = r & (T_ - 1);
  float ang = (float)t * __expf(-(float)d * (9.210340371976184f / 64.0f));
  float s, c;
  __sincosf(ang, &s, &c);
  u16* base = q + (size_t)r * ((size_t)NHEADS * HD_) + (size_t)h * HD_;
  float x1 = bf2f(base[d]);
  float x2 = bf2f(base[d + 64]);
  base[d]      = f2bf(x1 * c - x2 * s);
  base[d + 64] = f2bf(x2 * c + x1 * s);
}

// ---------------------------------------------------------------- GEMM
// C[M,N] = A[M,K] (bf16 row-major) * B[K,N] (bf16 row-major)
// 128x128 block tile, 8 waves, wave tile 32x64 (2x4 wmma tiles), k-tile 32.
// Double-buffered: A via async DMA, B via global->VGPR->LDS transpose,
// both for tile k+1 overlapping the 8 WMMAs of tile k.
#define GK 32
#define APITCH 40
#define BPITCH 40

template <int N, int K, bool F32OUT>
__global__ __launch_bounds__(256) void gemm_bf16(
    const u16* __restrict__ A, const u16* __restrict__ Bm,
    void* __restrict__ Cout, int M) {
  __shared__ u16 As[2][128 * APITCH];   // [row][k]
  __shared__ u16 Bt[2][128 * BPITCH];   // [col][k]  (B transposed)
  int tid = threadIdx.x;
  int lane = tid & 31, wave = tid >> 5;
  int wm = wave >> 1, wn = wave & 1;
  int m0 = blockIdx.y * 128, n0 = blockIdx.x * 128;

  // staging coordinates
  int ra_ = tid >> 1, ca_ = (tid & 1) * 16;       // A: 32B per thread
  int cb_ = (tid & 15) * 8, k2_ = (tid >> 4) * 2; // B: 2 k-rows x 8 cols
  const u16* Arow = A + (size_t)(m0 + ra_) * K + ca_;
  u32 aso0 = lds_off32(&As[0][ra_ * APITCH + ca_]);
  u32 aso1 = lds_off32(&As[1][ra_ * APITCH + ca_]);

  u32 breg[8];
  auto issueA = [&](int kt, int buf) {
    async_copy32(buf ? aso1 : aso0, Arow + kt * GK);
  };
  auto loadB = [&](int kt) {
    const u32* p0 = (const u32*)(Bm + (size_t)(kt * GK + k2_) * N + n0 + cb_);
    const u32* p1 = (const u32*)(Bm + (size_t)(kt * GK + k2_ + 1) * N + n0 + cb_);
#pragma unroll
    for (int u = 0; u < 4; ++u) { breg[u] = p0[u]; breg[4 + u] = p1[u]; }
  };
  auto storeB = [&](int buf) {
#pragma unroll
    for (int j = 0; j < 8; ++j) {
      u32 lo = (breg[j >> 1]     >> ((j & 1) * 16)) & 0xffffu;
      u32 hi = (breg[4 + (j >> 1)] >> ((j & 1) * 16)) & 0xffffu;
      *(u32*)(&Bt[buf][(cb_ + j) * BPITCH + k2_]) = lo | (hi << 16);
    }
  };

  v8f zero = {};
  v8f acc[2][4];
#pragma unroll
  for (int i = 0; i < 2; ++i)
#pragma unroll
    for (int j = 0; j < 4; ++j) acc[i][j] = zero;

  constexpr int NT = K / GK;
  issueA(0, 0);
  loadB(0);
  storeB(0);

  for (int kt = 0; kt < NT; ++kt) {
    int cur = kt & 1;
    if (kt + 1 < NT) {            // stream next tile while computing this one
      issueA(kt + 1, cur ^ 1);
      loadB(kt + 1);
      async_wait2();              // tile kt landed; kt+1 still in flight
    } else {
      async_wait0();
    }
    __syncthreads();              // tile kt visible to all waves

    const u16* as = As[cur];
    const u16* bs = Bt[cur];
    v16bf a0 = frag_ld(as, APITCH, wm * 32 + 0,  0, lane);
    v16bf a1 = frag_ld(as, APITCH, wm * 32 + 16, 0, lane);
#pragma unroll
    for (int cn = 0; cn < 4; ++cn) {
      v16bf b = frag_ld(bs, BPITCH, wn * 64 + cn * 16, 0, lane);
      acc[0][cn] = __builtin_amdgcn_wmma_f32_16x16x32_bf16(
          false, a0, false, b, (short)0, acc[0][cn], false, false);
      acc[1][cn] = __builtin_amdgcn_wmma_f32_16x16x32_bf16(
          false, a1, false, b, (short)0, acc[1][cn], false, false);
    }
    if (kt + 1 < NT) storeB(cur ^ 1);   // other buffer: race-free
    __syncthreads();              // all waves done reading tile kt
  }

  // epilogue: C/D layout row = i + 8*(lane/16), col = lane%16
  int g = lane >> 4, l16 = lane & 15;
  int col_base = n0 + wn * 64 + l16;
#pragma unroll
  for (int rm = 0; rm < 2; ++rm)
#pragma unroll
    for (int i = 0; i < 8; ++i) {
      int row = m0 + wm * 32 + rm * 16 + i + 8 * g;
      if (F32OUT) {
        float* cr = (float*)Cout + (size_t)row * N + col_base;
#pragma unroll
        for (int cn = 0; cn < 4; ++cn) cr[cn * 16] = acc[rm][cn][i];
      } else {
        u16* cr = (u16*)Cout + (size_t)row * N + col_base;
#pragma unroll
        for (int cn = 0; cn < 4; ++cn) cr[cn * 16] = f2bf(acc[rm][cn][i]);
      }
    }
}

// ---------------------------------------------------------------- attention
// One workgroup per (b, head, 64 query rows); 4 waves, 16 rows per wave.
#define BQ 64
#define BN 32
#define KPITCH 136
#define VPITCH 40
#define PPITCH 40

__global__ __launch_bounds__(128) void attn_kernel(
    const u16* __restrict__ Q, const u16* __restrict__ K,
    const u16* __restrict__ V, u16* __restrict__ O) {
  __shared__ u16 Qs[BQ * KPITCH];        // [qrow][d]
  __shared__ u16 Ks[BN * KPITCH];        // [key][d]
  __shared__ u16 Vt[HD_ * VPITCH];       // [d][key]   (V transposed)
  __shared__ u16 Ps[4 * 16 * PPITCH];    // per-wave P re-layout buffer

  int tid = threadIdx.x;
  int lane = tid & 31, wave = tid >> 5;
  int qt = blockIdx.x, h = blockIdx.y, b = blockIdx.z;
  int gkv = h >> 2;                        // NREP = 4
  int q0blk = qt * BQ;
  const u16* Qg = Q + (size_t)b * T_ * C_   + (size_t)h   * HD_;
  const u16* Kg = K + (size_t)b * T_ * KVD_ + (size_t)gkv * HD_;
  const u16* Vg = V + (size_t)b * T_ * KVD_ + (size_t)gkv * HD_;

  { // stage Q tile 64x128 via async DMA (128B contiguous per thread)
    int r = tid >> 1, c0 = (tid & 1) * 64;
    u32 so = lds_off32(&Qs[r * KPITCH + c0]);
    const u16* gp = Qg + (size_t)(q0blk + r) * C_ + c0;
    async_copy64(so, gp);
    async_copy64(so + 64, (const char*)gp + 64);
  }
  async_wait0();
  __syncthreads();
  v16bf qf[4];
#pragma unroll
  for (int kk = 0; kk < 4; ++kk)
    qf[kk] = frag_ld(Qs, KPITCH, wave * 16, kk * 32, lane);

  v8f zero = {};
  v8f oacc[8];
#pragma unroll
  for (int i = 0; i < 8; ++i) oacc[i] = zero;
  float mrow[8], lrow[8];
#pragma unroll
  for (int i = 0; i < 8; ++i) { mrow[i] = -1e30f; lrow[i] = 0.0f; }

  const float scale = 0.08838834764831845f;   // 1/sqrt(128)
  int grp = lane >> 4, l16 = lane & 15;
  int jmax = qt * 2 + 1;                      // causal upper key tile
  u16* Pw = &Ps[wave * 16 * PPITCH];

  for (int j = 0; j <= jmax; ++j) {
    int s0 = j * BN;
    __syncthreads();
    { // K tile 32x128 row-major: async 64B per thread
      int r = tid >> 2, c0 = (tid & 3) * 32;
      async_copy64(lds_off32(&Ks[r * KPITCH + c0]),
                   Kg + (size_t)(s0 + r) * KVD_ + c0);
    }
    { // V tile transposed: Vt[d][s], two keys per thread, packed b32 stores
      int s2 = (tid & 15) * 2;          // even key pair
      int d0 = (tid >> 4) * 16;         // column chunk
      const u32* p0 = (const u32*)(Vg + (size_t)(s0 + s2) * KVD_ + d0);
      const u32* p1 = (const u32*)(Vg + (size_t)(s0 + s2 + 1) * KVD_ + d0);
      u32 av[8], bv[8];
#pragma unroll
      for (int u = 0; u < 8; ++u) { av[u] = p0[u]; bv[u] = p1[u]; }
#pragma unroll
      for (int jj = 0; jj < 16; ++jj) {
        u32 lo = (av[jj >> 1] >> ((jj & 1) * 16)) & 0xffffu;
        u32 hi = (bv[jj >> 1] >> ((jj & 1) * 16)) & 0xffffu;
        *(u32*)(&Vt[(d0 + jj) * VPITCH + s2]) = lo | (hi << 16);
      }
    }
    if (j < jmax)   // prefetch next K tile
      __builtin_prefetch(Kg + (size_t)(s0 + BN) * KVD_, 0, 0);
    async_wait0();
    __syncthreads();

    // S = Q K^T : two 16x16 tiles, k-accumulate over HD=128
    v8f sfr[2];
#pragma unroll
    for (int jj = 0; jj < 2; ++jj) {
      v8f sa = zero;
#pragma unroll
      for (int kk = 0; kk < 4; ++kk) {
        v16bf bf = frag_ld(Ks, KPITCH, jj * 16, kk * 32, lane);
        sa = __builtin_amdgcn_wmma_f32_16x16x32_bf16(
            false, qf[kk], false, bf, (short)0, sa, false, false);
      }
      sfr[jj] = sa;
    }

    // online softmax (row lives in one 16-lane half -> xor 1/2/4/8 reduce)
#pragma unroll
    for (int i = 0; i < 8; ++i) {
      int trow = q0blk + wave * 16 + i + 8 * grp;
      float v0 = sfr[0][i] * scale;
      float v1 = sfr[1][i] * scale;
      if (s0 + l16      > trow) v0 = -1e30f;
      if (s0 + 16 + l16 > trow) v1 = -1e30f;
      float mx = fmaxf(v0, v1);
      mx = fmaxf(mx, __shfl_xor(mx, 1, 32));
      mx = fmaxf(mx, __shfl_xor(mx, 2, 32));
      mx = fmaxf(mx, __shfl_xor(mx, 4, 32));
      mx = fmaxf(mx, __shfl_xor(mx, 8, 32));
      float mnew = fmaxf(mrow[i], mx);
      float alpha = __expf(mrow[i] - mnew);
      lrow[i] *= alpha;
#pragma unroll
      for (int dt = 0; dt < 8; ++dt) oacc[dt][i] *= alpha;
      float p0 = __expf(v0 - mnew);
      float p1 = __expf(v1 - mnew);
      float ls = p0 + p1;
      ls += __shfl_xor(ls, 1, 32);
      ls += __shfl_xor(ls, 2, 32);
      ls += __shfl_xor(ls, 4, 32);
      ls += __shfl_xor(ls, 8, 32);
      lrow[i] += ls;
      mrow[i] = mnew;
      int mq = i + 8 * grp;                 // re-layout P -> A-fragment form
      Pw[mq * PPITCH + l16]      = f2bf(p0);
      Pw[mq * PPITCH + 16 + l16] = f2bf(p1);
    }
    asm volatile("s_wait_dscnt 0" ::: "memory");

    // O += P V : A = P (16x32 keys), B = V tile column blocks
    v16bf pf = frag_ld(Pw, PPITCH, 0, 0, lane);
#pragma unroll
    for (int dt = 0; dt < 8; ++dt) {
      v16bf vf = frag_ld(Vt, VPITCH, dt * 16, 0, lane);
      oacc[dt] = __builtin_amdgcn_wmma_f32_16x16x32_bf16(
          false, pf, false, vf, (short)0, oacc[dt], false, false);
    }
  }

  // normalize + store (bf16, row-major [B*T, C])
#pragma unroll
  for (int i = 0; i < 8; ++i) {
    float inv = 1.0f / lrow[i];
    int trow = q0blk + wave * 16 + i + 8 * grp;
    u16* orow = O + (size_t)(b * T_ + trow) * C_ + (size_t)h * HD_;
#pragma unroll
    for (int dt = 0; dt < 8; ++dt)
      orow[dt * 16 + l16] = f2bf(oacc[dt][i] * inv);
  }
}

// ---------------------------------------------------------------- launch
extern "C" void kernel_launch(void* const* d_in, const int* in_sizes, int n_in,
                              void* d_out, int out_size, void* d_ws, size_t ws_size,
                              hipStream_t stream) {
  const float* x  = (const float*)d_in[0];
  const float* Wq = (const float*)d_in[1];
  const float* Wk = (const float*)d_in[2];
  const float* Wv = (const float*)d_in[3];
  const float* Wo = (const float*)d_in[4];

  const size_t NTOK = (size_t)B_ * T_;       // 4096 rows
  u16* p = (u16*)d_ws;
  u16* xb  = p; p += NTOK * C_;
  u16* Wqb = p; p += (size_t)C_ * C_;
  u16* Wkb = p; p += (size_t)C_ * KVD_;
  u16* Wvb = p; p += (size_t)C_ * KVD_;
  u16* Wob = p; p += (size_t)C_ * C_;
  u16* Qb  = p; p += NTOK * C_;
  u16* Kb  = p; p += NTOK * KVD_;
  u16* Vb  = p; p += NTOK * KVD_;
  u16* Ob  = p; p += NTOK * C_;

  auto cvt = [&](const float* s, u16* d, size_t n) {
    cvt_f32_bf16<<<dim3((unsigned)((n / 4 + 255) / 256)), 256, 0, stream>>>(
        (const float4*)s, (uint2*)d, (int)(n / 4));
  };
  cvt(x,  xb,  NTOK * C_);
  cvt(Wq, Wqb, (size_t)C_ * C_);
  cvt(Wk, Wkb, (size_t)C_ * KVD_);
  cvt(Wv, Wvb, (size_t)C_ * KVD_);
  cvt(Wo, Wob, (size_t)C_ * C_);

  // QKV projections
  gemm_bf16<C_, C_, false><<<dim3(C_ / 128, NTOK / 128), 256, 0, stream>>>(
      xb, Wqb, Qb, (int)NTOK);
  gemm_bf16<KVD_, C_, false><<<dim3(KVD_ / 128, NTOK / 128), 256, 0, stream>>>(
      xb, Wkb, Kb, (int)NTOK);
  gemm_bf16<KVD_, C_, false><<<dim3(KVD_ / 128, NTOK / 128), 256, 0, stream>>>(
      xb, Wvb, Vb, (int)NTOK);

  // RoPE on Q and K
  rope_kernel<NH_><<<dim3((B_ * T_ * NH_ * 64 + 255) / 256), 256, 0, stream>>>(Qb);
  rope_kernel<NKV_><<<dim3((B_ * T_ * NKV_ * 64 + 255) / 256), 256, 0, stream>>>(Kb);

  // causal GQA flash attention
  attn_kernel<<<dim3(T_ / BQ, NH_, B_), 128, 0, stream>>>(Qb, Kb, Vb, Ob);

  // output projection (f32 out)
  gemm_bf16<C_, C_, true><<<dim3(C_ / 128, NTOK / 128), 256, 0, stream>>>(
      Ob, Wob, (float*)d_out, (int)NTOK);
}
